// ViTMAEAttentionBlock_16982300689095
// MI455X (gfx1250) — matrix operand order repaired and checked by reading, not compiled, not run on hardware
//
#include <hip/hip_runtime.h>
#include <hip/hip_bf16.h>
#include <stdint.h>

#define HIDDEN   1024
#define HEADS    16
#define HEAD_DIM 64
#define BATCH    16
#define SEQ      1024
#define MTOT     (BATCH * SEQ)   // 16384

typedef __bf16 bf16;
typedef __attribute__((ext_vector_type(16))) __bf16    v16bf;
typedef __attribute__((ext_vector_type(8)))  float     v8f;
typedef __attribute__((ext_vector_type(4)))  uint32_t  u32x4;
typedef __attribute__((ext_vector_type(8)))  uint32_t  u32x8;

union Frag {
    uint4 u[2];
    v16bf v;
};

__device__ __forceinline__ uint4 ld_u4(const void* p) {
    uint4 r; __builtin_memcpy(&r, p, 16); return r;
}
__device__ __forceinline__ v8f wmma_bf16(const Frag& a, const Frag& b, v8f c) {
    return __builtin_amdgcn_wmma_f32_16x16x32_bf16(false, a.v, false, b.v,
                                                   (short)0, c, false, false);
}

// LDS byte offset of a __shared__ object: flat LDS aperture keeps the
// workgroup-relative offset in addr[31:0].
__device__ __forceinline__ uint32_t lds_off(const void* p) {
    return (uint32_t)(uintptr_t)p;
}

// ---------------------------------------------------------------------------
// Tensor Data Mover: DMA a 2D tile (tile_w elems x tile_h rows, 2-byte
// elements, row stride stride_elems) from global memory into LDS at lds_byte.
// Descriptor packing per CDNA5 ISA sect. 8.3/8.4.
// Tracked by TENSORcnt.
// ---------------------------------------------------------------------------
__device__ __forceinline__ void tdm_load_2d(uint32_t lds_byte, const void* gptr,
                                            uint32_t tile_w, uint32_t tile_h,
                                            uint32_t stride_elems)
{
    uint64_t ga = (uint64_t)(uintptr_t)gptr;
    u32x4 g0;
    g0[0] = 1u;                                             // count=1, user mode
    g0[1] = lds_byte;                                       // lds_addr [63:32]
    g0[2] = (uint32_t)ga;                                   // global_addr lo
    g0[3] = (uint32_t)((ga >> 32) & 0x01FFFFFFu)            // global_addr[56:32]
          | 0x80000000u;                                    // type=2 ("image")
    u32x8 g1;
    g1[0] = 0x00010000u;                                    // data_size=1 (2B)
    g1[1] = (tile_w & 0xFFFFu) << 16;                       // tensor_dim0 lo16
    g1[2] = (tile_w >> 16) | ((tile_h & 0xFFFFu) << 16);    // dim0 hi | dim1 lo
    g1[3] = (tile_h >> 16) | (tile_w << 16);                // dim1 hi | tile_dim0
    g1[4] = tile_h;                                         // tile_dim1 (dim2=0)
    g1[5] = stride_elems;                                   // dim0_stride lo32
    g1[6] = 0u;                                             // stride hi | dim1_stride lo
    g1[7] = 0u;
    asm volatile("tensor_load_to_lds %0, %1" :: "s"(g0), "s"(g1) : "memory");
}

// ---------------------------------------------------------------------------
// fp32 -> bf16 conversion (grid-stride)
// ---------------------------------------------------------------------------
__global__ void cvt_f32_bf16(const float* __restrict__ src, bf16* __restrict__ dst, int n) {
    int i = blockIdx.x * blockDim.x + threadIdx.x;
    int stride = gridDim.x * blockDim.x;
    for (; i < n; i += stride) dst[i] = (bf16)src[i];
}

// ---------------------------------------------------------------------------
// Weight transpose + convert: W[K,N] f32 -> Wt[N,K] bf16 (32x32 LDS tiles)
// so GEMM B tiles become plain 2D strided copies (TDM friendly, and the
// LDS Bts[n][k] layout matches the WMMA B fragment pair packing).
// ---------------------------------------------------------------------------
__global__ __launch_bounds__(256)
void transpose_cvt(const float* __restrict__ W, bf16* __restrict__ Wt) {
    __shared__ float tile[32][33];
    const int n0 = blockIdx.x * 32, k0 = blockIdx.y * 32;
    const int tx = threadIdx.x & 31, ty = threadIdx.x >> 5;   // 8 rows/pass
    #pragma unroll
    for (int i = 0; i < 4; ++i) {
        int k = ty + i * 8;
        tile[k][tx] = W[(size_t)(k0 + k) * HIDDEN + n0 + tx];
    }
    __syncthreads();
    #pragma unroll
    for (int i = 0; i < 4; ++i) {
        int n = ty + i * 8;
        Wt[(size_t)(n0 + n) * HIDDEN + k0 + tx] = (bf16)tile[tx][n];
    }
}

// ---------------------------------------------------------------------------
// GEMM: Y[M,N] = A[M,K] @ Wt[N,K]^T + bias,  M=16384, K=N=1024
// Block tile 128x64, K-step 32, TDM double-buffered LDS staging.
// 256 threads = 8 waves; each wave computes a 16x64 strip as four 16x16
// f32 WMMA accumulators.
// mode 0: bf16 out in [B, H, S, D] (split heads);  mode 1: fp32 [M, N].
// ---------------------------------------------------------------------------
__global__ __launch_bounds__(256)
void gemm_bf16(const bf16* __restrict__ A,
               const bf16* __restrict__ Wt,     // [N, K] row-major
               const float* __restrict__ bias,
               bf16* __restrict__ outSplit,
               float* __restrict__ outDense,
               int mode)
{
    __shared__ __align__(16) bf16 Asm[2][128 * 32];   // A tile [row][k], x2
    __shared__ __align__(16) bf16 Bts[2][64 * 32];    // W tile [n][k],  x2

    const int tid  = threadIdx.x;
    const int lane = tid & 31;
    const int wave = tid >> 5;         // 0..7 -> 16 M-rows each
    const int half = lane >> 4;        // 0/1
    const int lr   = lane & 15;
    const int m0   = blockIdx.x * 128;
    const int n0   = blockIdx.y * 64;

    const uint32_t offA[2] = { lds_off(&Asm[0][0]), lds_off(&Asm[1][0]) };
    const uint32_t offB[2] = { lds_off(&Bts[0][0]), lds_off(&Bts[1][0]) };
    const bf16* Abase = A  + (size_t)m0 * HIDDEN;
    const bf16* Bbase = Wt + (size_t)n0 * HIDDEN;

    const v8f vzero = {0.f,0.f,0.f,0.f,0.f,0.f,0.f,0.f};
    v8f acc[4];
    #pragma unroll
    for (int i = 0; i < 4; ++i) acc[i] = vzero;

    // prologue: DMA first tiles into buffer 0
    if (wave == 0) {
        tdm_load_2d(offA[0], Abase, 32, 128, HIDDEN);
        tdm_load_2d(offB[0], Bbase, 32,  64, HIDDEN);
    }

    int buf = 0;
    for (int k0 = 0; k0 < HIDDEN; k0 += 32, buf ^= 1) {
        if (wave == 0) __builtin_amdgcn_s_wait_tensorcnt(0);  // tiles landed
        __syncthreads();
        // async prefetch of next K-step into the other buffer
        if (wave == 0 && (k0 + 32) < HIDDEN) {
            tdm_load_2d(offA[buf ^ 1], Abase + k0 + 32, 32, 128, HIDDEN);
            tdm_load_2d(offB[buf ^ 1], Bbase + k0 + 32, 32,  64, HIDDEN);
        }

        const bf16* As = Asm[buf];
        const bf16* Bs = Bts[buf];
        Frag af;
        const bf16* arow = As + (wave * 16 + lr) * 32;
        af.u[0] = ld_u4(arow + half * 8);        // K 0..7  / 8..15
        af.u[1] = ld_u4(arow + 16 + half * 8);   // K 16..23 / 24..31
        #pragma unroll
        for (int ns = 0; ns < 4; ++ns) {
            Frag bfr;
            const bf16* brow = Bs + (ns * 16 + lr) * 32;
            bfr.u[0] = ld_u4(brow + half * 8);       // K 0..7  / 8..15
            bfr.u[1] = ld_u4(brow + 16 + half * 8);  // K 16..23 / 24..31
            acc[ns] = wmma_bf16(af, bfr, acc[ns]);
        }
        __syncthreads();   // everyone done reading buf before TDM reuses it
    }

    // --- epilogue: bias + store (C layout: row = 8*half + r, col = lr) ---
    #pragma unroll
    for (int ns = 0; ns < 4; ++ns) {
        int col = n0 + ns * 16 + lr;
        float b = bias[col];
        #pragma unroll
        for (int r = 0; r < 8; ++r) {
            int row = m0 + wave * 16 + half * 8 + r;
            float v = acc[ns][r] + b;
            if (mode == 0) {
                int bi = row >> 10, srow = row & (SEQ - 1);
                int h  = col >> 6,  d    = col & (HEAD_DIM - 1);
                outSplit[(((size_t)(bi * HEADS + h)) * SEQ + srow) * HEAD_DIM + d] = (bf16)v;
            } else {
                outDense[(size_t)row * HIDDEN + col] = v;
            }
        }
    }
}

// ---------------------------------------------------------------------------
// Flash attention: one block per (64 queries, batch*head). 128 threads =
// 4 waves x 16 queries. Streams keys/values in 32-wide tiles with online
// softmax; P@V via bf16 WMMA with LDS-transposed V.
// ---------------------------------------------------------------------------
__global__ __launch_bounds__(128)
void attention(const bf16* __restrict__ Q,   // [B*H, S, D]
               const bf16* __restrict__ K,
               const bf16* __restrict__ V,
               bf16* __restrict__ Ctx)       // [B, S, HIDDEN]
{
    __shared__ __align__(16) bf16 Vt[64 * 32];       // V transposed [d][key]
    __shared__ __align__(16) bf16 Pl[4 * 16 * 32];   // per-wave P scratch [16][32]

    const int tid  = threadIdx.x;
    const int lane = tid & 31;
    const int wave = tid >> 5;       // 0..3
    const int half = lane >> 4;
    const int lr   = lane & 15;
    const int qt   = blockIdx.x;     // query tile (64 rows)
    const int bh   = blockIdx.y;     // batch*head

    const bf16* Qb = Q + (size_t)bh * SEQ * HEAD_DIM;
    const bf16* Kb = K + (size_t)bh * SEQ * HEAD_DIM;
    const bf16* Vb = V + (size_t)bh * SEQ * HEAD_DIM;

    // Q fragments for this wave's 16 query rows (row-contiguous D in memory)
    Frag qf[2];
    {
        const bf16* qr = Qb + (size_t)(qt * 64 + wave * 16 + lr) * HEAD_DIM;
        qf[0].u[0] = ld_u4(qr + half * 8);
        qf[0].u[1] = ld_u4(qr + 16 + half * 8);
        qf[1].u[0] = ld_u4(qr + 32 + half * 8);
        qf[1].u[1] = ld_u4(qr + 48 + half * 8);
    }

    const v8f vzero = {0.f,0.f,0.f,0.f,0.f,0.f,0.f,0.f};
    v8f oacc[4];
    #pragma unroll
    for (int i = 0; i < 4; ++i) oacc[i] = vzero;
    float m[8], s[8];
    #pragma unroll
    for (int r = 0; r < 8; ++r) { m[r] = -3.0e38f; s[r] = 0.f; }

    const float scale = 0.125f;   // 1/sqrt(64)

    for (int kb = 0; kb < SEQ; kb += 32) {
        if (kb + 32 < SEQ) {  // prefetch next tiles into cache
            __builtin_prefetch(Kb + (size_t)(kb + 32 + (tid & 31)) * HEAD_DIM, 0, 1);
            __builtin_prefetch(Vb + (size_t)(kb + 32 + (tid & 31)) * HEAD_DIM, 0, 1);
        }
        __syncthreads();   // previous iteration's Vt reads complete
        // cooperative transpose of V tile [32 keys][64 d] -> Vt[d][key]
        #pragma unroll
        for (int i = 0; i < 4; ++i) {
            int key = (tid >> 4) + i * 8;     // 0..31
            int d0  = (tid & 15) * 4;
            uint2 dv;
            __builtin_memcpy(&dv, Vb + (size_t)(kb + key) * HEAD_DIM + d0, 8);
            bf16 t4[4];
            __builtin_memcpy(t4, &dv, 8);
            #pragma unroll
            for (int j = 0; j < 4; ++j) Vt[(d0 + j) * 32 + key] = t4[j];
        }
        __syncthreads();

        // scores for two 16-key subtiles, online softmax, stage P in LDS
        #pragma unroll
        for (int ks = 0; ks < 2; ++ks) {
            v8f sacc = vzero;
            const bf16* krow = Kb + (size_t)(kb + ks * 16 + lr) * HEAD_DIM;
            Frag kf;
            kf.u[0] = ld_u4(krow + half * 8);         // d 0..7 / 8..15
            kf.u[1] = ld_u4(krow + 16 + half * 8);    // d 16..23 / 24..31
            sacc = wmma_bf16(qf[0], kf, sacc);
            kf.u[0] = ld_u4(krow + 32 + half * 8);    // d 32..39 / 40..47
            kf.u[1] = ld_u4(krow + 48 + half * 8);
            sacc = wmma_bf16(qf[1], kf, sacc);

            #pragma unroll
            for (int r = 0; r < 8; ++r) {
                float sv = sacc[r] * scale;
                float tm = sv;
                tm = fmaxf(tm, __shfl_xor(tm, 1));
                tm = fmaxf(tm, __shfl_xor(tm, 2));
                tm = fmaxf(tm, __shfl_xor(tm, 4));
                tm = fmaxf(tm, __shfl_xor(tm, 8));
                float mn   = fmaxf(m[r], tm);
                float corr = __expf(m[r] - mn);
                float p    = __expf(sv - mn);
                float ps   = p;
                ps += __shfl_xor(ps, 1);
                ps += __shfl_xor(ps, 2);
                ps += __shfl_xor(ps, 4);
                ps += __shfl_xor(ps, 8);
                s[r] = s[r] * corr + ps;
                m[r] = mn;
                #pragma unroll
                for (int nc = 0; nc < 4; ++nc) oacc[nc][r] *= corr;
                Pl[wave * 512 + (half * 8 + r) * 32 + ks * 16 + lr] = (bf16)p;
            }
        }

        // O += P(16x32) @ V(32x64)
        Frag pf;
        const bf16* prow = Pl + wave * 512 + lr * 32;
        pf.u[0] = ld_u4(prow + half * 8);
        pf.u[1] = ld_u4(prow + 16 + half * 8);
        #pragma unroll
        for (int nc = 0; nc < 4; ++nc) {
            Frag vf;
            const bf16* vrow = Vt + (nc * 16 + lr) * 32;
            vf.u[0] = ld_u4(vrow + half * 8);
            vf.u[1] = ld_u4(vrow + 16 + half * 8);
            oacc[nc] = wmma_bf16(pf, vf, oacc[nc]);
        }
    }

    // normalize and write ctx in [B, S, HIDDEN] (merged heads), bf16
    const int b = bh >> 4, h = bh & (HEADS - 1);
    #pragma unroll
    for (int r = 0; r < 8; ++r) {
        int   srow = qt * 64 + wave * 16 + half * 8 + r;
        float rinv = 1.0f / s[r];
        #pragma unroll
        for (int nc = 0; nc < 4; ++nc) {
            float v = oacc[nc][r] * rinv;
            Ctx[((size_t)(b * SEQ + srow)) * HIDDEN + h * HEAD_DIM + nc * 16 + lr] = (bf16)v;
        }
    }
}

// ---------------------------------------------------------------------------
extern "C" void kernel_launch(void* const* d_in, const int* in_sizes, int n_in,
                              void* d_out, int out_size, void* d_ws, size_t ws_size,
                              hipStream_t stream)
{
    (void)in_sizes; (void)n_in; (void)out_size; (void)ws_size;
    const float* x  = (const float*)d_in[0];
    const float* Wq = (const float*)d_in[1];
    const float* bq = (const float*)d_in[2];
    const float* Wk = (const float*)d_in[3];
    const float* bk = (const float*)d_in[4];
    const float* Wv = (const float*)d_in[5];
    const float* bv = (const float*)d_in[6];
    const float* Wd = (const float*)d_in[7];
    const float* bd = (const float*)d_in[8];
    float* out = (float*)d_out;

    char* ws = (char*)d_ws;
    size_t off = 0;
    bf16* xb   = (bf16*)(ws + off); off += (size_t)MTOT * HIDDEN * 2;     // 32 MB
    bf16* wqb  = (bf16*)(ws + off); off += (size_t)HIDDEN * HIDDEN * 2;   //  2 MB (transposed)
    bf16* wkb  = (bf16*)(ws + off); off += (size_t)HIDDEN * HIDDEN * 2;
    bf16* wvb  = (bf16*)(ws + off); off += (size_t)HIDDEN * HIDDEN * 2;
    bf16* wdb  = (bf16*)(ws + off); off += (size_t)HIDDEN * HIDDEN * 2;
    bf16* Qb   = (bf16*)(ws + off); off += (size_t)MTOT * HIDDEN * 2;     // 32 MB
    bf16* Kb   = (bf16*)(ws + off); off += (size_t)MTOT * HIDDEN * 2;
    bf16* Vb   = (bf16*)(ws + off); off += (size_t)MTOT * HIDDEN * 2;
    bf16* Ctxb = (bf16*)(ws + off); off += (size_t)MTOT * HIDDEN * 2;

    cvt_f32_bf16<<<4096, 256, 0, stream>>>(x, xb, MTOT * HIDDEN);
    dim3 gt(HIDDEN / 32, HIDDEN / 32);
    transpose_cvt<<<gt, 256, 0, stream>>>(Wq, wqb);
    transpose_cvt<<<gt, 256, 0, stream>>>(Wk, wkb);
    transpose_cvt<<<gt, 256, 0, stream>>>(Wv, wvb);
    transpose_cvt<<<gt, 256, 0, stream>>>(Wd, wdb);

    dim3 gg(MTOT / 128, HIDDEN / 64);
    gemm_bf16<<<gg, 256, 0, stream>>>(xb, wqb, bq, Qb, nullptr, 0);
    gemm_bf16<<<gg, 256, 0, stream>>>(xb, wkb, bk, Kb, nullptr, 0);
    gemm_bf16<<<gg, 256, 0, stream>>>(xb, wvb, bv, Vb, nullptr, 0);

    dim3 ga(SEQ / 64, BATCH * HEADS);
    attention<<<ga, 128, 0, stream>>>(Qb, Kb, Vb, Ctxb);

    gemm_bf16<<<gg, 256, 0, stream>>>(Ctxb, wdb, bd, nullptr, out, 1);
}
